// AppearanceBlock_25984552141309
// MI455X (gfx1250) — compile-verified, matrix operand order checked by reading, not run on hardware
//
#include <hip/hip_runtime.h>
#include <cstdint>

typedef __attribute__((ext_vector_type(16))) _Float16 v16h;
typedef __attribute__((ext_vector_type(8)))  _Float16 v8h;
typedef __attribute__((ext_vector_type(2)))  _Float16 v2h;
typedef __attribute__((ext_vector_type(8)))  float    v8f;
typedef __attribute__((ext_vector_type(4)))  float    v4f;
typedef __attribute__((ext_vector_type(4)))  int      v4i;

#define B_   4
#define C_   256
#define CQ   32
#define N_   4096
#define WPB  8            // waves per attention block (256 threads)

// ---------------------------------------------------------------------------
// CDNA5 async global->LDS copy (ASYNCcnt-tracked), guarded so a missing
// builtin degrades to a plain load + ds_store without breaking the build.
// Builtin expects (int4 AS1*, int4 AS3*, imm offset, imm cpol) per the
// clang diagnostic ("int4 __device__ *").
// ---------------------------------------------------------------------------
#if __has_builtin(__builtin_amdgcn_global_load_async_to_lds_b128)
#define USE_ASYNC_LDS 1
typedef __attribute__((address_space(1))) v4i gv4i;
typedef __attribute__((address_space(3))) v4i lv4i;
#endif

__device__ __forceinline__ void cp16_async(void* lds, const void* g) {
#ifdef USE_ASYNC_LDS
  __builtin_amdgcn_global_load_async_to_lds_b128(
      (gv4i*)(uintptr_t)g, (lv4i*)(unsigned)(uintptr_t)lds, 0, 0);
#else
  *(v8h*)lds = *(const v8h*)g;
#endif
}

__device__ __forceinline__ void wait_async_all() {
#ifdef USE_ASYNC_LDS
#if __has_builtin(__builtin_amdgcn_s_wait_asynccnt)
  __builtin_amdgcn_s_wait_asynccnt(0);
#else
  asm volatile("s_wait_asynccnt 0" ::: "memory");
#endif
#endif
}

// K-slot permutation for the PV matmul (shared by P-store and vF layout):
//   K slot j <-> n offset (j>>1) | ((j&1)<<4)   within each 32-column chunk.

// ---------------------------------------------------------------------------
// Kernel 1: 1x1-conv projections.
//   q = Wq @ pose + bq   -> qT [B][N][CQ] f16   (K-contiguous for WMMA gather)
//   k = Wk @ src  + bk   -> kT [B][N][CQ] f16
//   v = Wv @ src  + bv   -> vF [B][C][N]  f16   (n pre-permuted per 32-chunk)
// ---------------------------------------------------------------------------
__global__ __launch_bounds__(256)
void proj_kernel(const float* __restrict__ source, const float* __restrict__ pose,
                 const float* __restrict__ Wq, const float* __restrict__ bq,
                 const float* __restrict__ Wk, const float* __restrict__ bk,
                 const float* __restrict__ Wv, const float* __restrict__ bv,
                 _Float16* __restrict__ qT, _Float16* __restrict__ kT,
                 _Float16* __restrict__ vF) {
  __shared__ __align__(16) _Float16 sh[C_ * 64];   // 32 KB tile [c][n]
  const int tid = threadIdx.x;
  const int blk = blockIdx.x;
  const int b   = blk / (N_ / 64);
  const int n0  = (blk % (N_ / 64)) * 64;
  const int n   = tid & 63;
  const int grp = tid >> 6;            // uniform within a wave -> scalar weight loads

  for (int idx = tid; idx < C_ * 64; idx += 256) {
    int c = idx >> 6, j = idx & 63;
    sh[idx] = (_Float16)pose[((size_t)b * C_ + c) * N_ + n0 + j];
  }
  __syncthreads();
  for (int oo = 0; oo < 8; ++oo) {
    const int o = grp * 8 + oo;
    const float* __restrict__ w = Wq + o * C_;
    float acc = bq[o];
    #pragma unroll 8
    for (int c = 0; c < C_; ++c) acc += w[c] * (float)sh[c * 64 + n];
    qT[((size_t)b * N_ + n0 + n) * CQ + o] = (_Float16)acc;
  }
  __syncthreads();

  for (int idx = tid; idx < C_ * 64; idx += 256) {
    int c = idx >> 6, j = idx & 63;
    sh[idx] = (_Float16)source[((size_t)b * C_ + c) * N_ + n0 + j];
  }
  __syncthreads();
  for (int oo = 0; oo < 8; ++oo) {
    const int o = grp * 8 + oo;
    const float* __restrict__ w = Wk + o * C_;
    float acc = bk[o];
    #pragma unroll 8
    for (int c = 0; c < C_; ++c) acc += w[c] * (float)sh[c * 64 + n];
    kT[((size_t)b * N_ + n0 + n) * CQ + o] = (_Float16)acc;
  }
  // inverse K-slot permutation for v: column x (0..31) -> slot ((x&15)<<1)|(x>>4)
  const int x  = n & 31;
  const int jp = ((x & 15) << 1) | (x >> 4);
  const size_t vcol = (size_t)(n0 + (n & 32) + jp);
  for (int oo = 0; oo < 64; ++oo) {
    const int o = grp * 64 + oo;
    const float* __restrict__ w = Wv + o * C_;
    float acc = bv[o];
    #pragma unroll 8
    for (int c = 0; c < C_; ++c) acc += w[c] * (float)sh[c * 64 + n];
    vF[((size_t)b * C_ + o) * N_ + vcol] = (_Float16)acc;
  }
}

// ---------------------------------------------------------------------------
// Kernel 2: fused two-pass flash attention.
// 8 waves/block, all sharing one batch b; each wave owns a 16-row m-tile.
// Pass B double-buffers the shared K/V chunk in LDS via async global->LDS
// copies (one fetch per block instead of one per wave: 8x less L2 traffic).
// ---------------------------------------------------------------------------
__global__ __launch_bounds__(256)
void attn_kernel(const _Float16* __restrict__ qT, const _Float16* __restrict__ kT,
                 const _Float16* __restrict__ vF, const float* __restrict__ src,
                 const float* __restrict__ gamma_p, float* __restrict__ out) {
  __shared__ __align__(16) _Float16 shV[2][C_ * 32];   // 2 x 16 KB V chunk [c][slot]
  __shared__ __align__(16) _Float16 shK[2][32 * 32];   // 2 x  2 KB K chunk [n][o]
  __shared__ __align__(64) _Float16 shP[WPB][16 * 32]; // per-wave P transpose

  const int tid  = threadIdx.x;
  const int lane = tid & 31;
  const int wv   = tid >> 5;
  const int b    = blockIdx.x / (N_ / 16 / WPB);               // uniform per block
  const int m0   = ((blockIdx.x % (N_ / 16 / WPB)) * WPB + wv) * 16;
  const int half = lane >> 4;           // 0: rows 0-7/K 0-7,16-23 ; 1: rows 8-15/K 8-15,24-31
  const int lr   = lane & 15;
  const int koff = half ? 8 : 0;

  // A-operand: q rows m0..m0+15, K-packed per ISA 16-bit A layout
  const _Float16* qrow = qT + ((size_t)b * N_ + m0 + lr) * CQ;
  const v8h qlo = *(const v8h*)(qrow + koff);
  const v8h qhi = *(const v8h*)(qrow + koff + 16);
  const v16h aq = __builtin_shufflevector(qlo, qhi, 0,1,2,3,4,5,6,7,8,9,10,11,12,13,14,15);

  const _Float16* kbase = kT + ((size_t)b * N_ + lr) * CQ;
  const v8f czero = {};

  // ================= Pass A: exact row max (per-lane accumulation) ==========
  float rmax[8];
  #pragma unroll
  for (int i = 0; i < 8; ++i) rmax[i] = -3.0e38f;

  for (int n0 = 0; n0 < N_; n0 += 32) {
    const _Float16* kr0 = kbase + (size_t)n0 * CQ;
    const _Float16* kr1 = kr0 + 16 * CQ;
    const v8h kl0 = *(const v8h*)(kr0 + koff);
    const v8h kh0 = *(const v8h*)(kr0 + koff + 16);
    const v8h kl1 = *(const v8h*)(kr1 + koff);
    const v8h kh1 = *(const v8h*)(kr1 + koff + 16);
    const v16h kb0 = __builtin_shufflevector(kl0, kh0, 0,1,2,3,4,5,6,7,8,9,10,11,12,13,14,15);
    const v16h kb1 = __builtin_shufflevector(kl1, kh1, 0,1,2,3,4,5,6,7,8,9,10,11,12,13,14,15);
    __builtin_prefetch(kr0 + 32 * CQ, 0, 3);   // near-scope prefetch

    const v8f e0 = __builtin_amdgcn_wmma_f32_16x16x32_f16(false, aq, false, kb0,
                                                          (short)0, czero, false, false);
    const v8f e1 = __builtin_amdgcn_wmma_f32_16x16x32_f16(false, aq, false, kb1,
                                                          (short)0, czero, false, false);
    #pragma unroll
    for (int i = 0; i < 8; ++i) rmax[i] = fmaxf(rmax[i], fmaxf(e0[i], e1[i]));
  }
  #pragma unroll
  for (int i = 0; i < 8; ++i) {   // one-time 16-lane reductions
    float m = rmax[i];
    m = fmaxf(m, __shfl_xor(m, 1, 32));
    m = fmaxf(m, __shfl_xor(m, 2, 32));
    m = fmaxf(m, __shfl_xor(m, 4, 32));
    m = fmaxf(m, __shfl_xor(m, 8, 32));
    rmax[i] = m;
  }

  // ================= Pass B: P = exp(e - max), PV accumulation ==============
  v8f acc[16];
  #pragma unroll
  for (int t = 0; t < 16; ++t)
    #pragma unroll
    for (int i = 0; i < 8; ++i) acc[t][i] = 0.f;
  float rsum[8];
  #pragma unroll
  for (int i = 0; i < 8; ++i) rsum[i] = 0.f;

  _Float16* myP = &shP[wv][0];
  const _Float16* vbase = vF + (size_t)b * C_ * N_;

  // cooperative staging roles: 256 threads, 16B per async op
  const int part = tid & 3;      // 16B chunk within a 64B row
  const int crow = tid >> 2;     // 0..63

  auto stage = [&](int buf, int n0) {
    #pragma unroll
    for (int r = 0; r < 4; ++r) {
      const int c = crow + 64 * r;
      cp16_async(&shV[buf][c * 32 + part * 8],
                 vbase + (size_t)c * N_ + n0 + part * 8);
    }
    if (tid < 128) {             // 32 rows x 32 halves of K
      cp16_async(&shK[buf][crow * 32 + part * 8],
                 kT + ((size_t)b * N_ + n0 + crow) * CQ + part * 8);
    }
  };

  stage(0, 0);
  wait_async_all();
  __syncthreads();

  for (int ch = 0; ch < N_ / 32; ++ch) {
    const int cur = ch & 1;
    if (ch + 1 < N_ / 32) stage(cur ^ 1, (ch + 1) * 32);

    // ---- energy from staged K ----
    const _Float16* kk = &shK[cur][0];
    const v8h kl0 = *(const v8h*)(kk + lr * 32 + koff);
    const v8h kh0 = *(const v8h*)(kk + lr * 32 + koff + 16);
    const v8h kl1 = *(const v8h*)(kk + (16 + lr) * 32 + koff);
    const v8h kh1 = *(const v8h*)(kk + (16 + lr) * 32 + koff + 16);
    const v16h kb0 = __builtin_shufflevector(kl0, kh0, 0,1,2,3,4,5,6,7,8,9,10,11,12,13,14,15);
    const v16h kb1 = __builtin_shufflevector(kl1, kh1, 0,1,2,3,4,5,6,7,8,9,10,11,12,13,14,15);

    const v8f e0 = __builtin_amdgcn_wmma_f32_16x16x32_f16(false, aq, false, kb0,
                                                          (short)0, czero, false, false);
    const v8f e1 = __builtin_amdgcn_wmma_f32_16x16x32_f16(false, aq, false, kb1,
                                                          (short)0, czero, false, false);

    // ---- P tile: packed (p0,p1) -> one b32 LDS store per row slot ----
    // Stored slot j maps to n offset (j>>1)|((j&1)<<4); vF uses same mapping.
    #pragma unroll
    for (int i = 0; i < 8; ++i) {
      const float a0 = __expf(e0[i] - rmax[i]);
      const float a1 = __expf(e1[i] - rmax[i]);
      rsum[i] += a0 + a1;                    // per-lane; reduced after the loop
      const int r = i + 8 * half;
      v2h pk; pk[0] = (_Float16)a0; pk[1] = (_Float16)a1;
      *(v2h*)(myP + r * 32 + 2 * lr) = pk;
    }
    __builtin_amdgcn_wave_barrier();         // per-wave LDS is in-order
    const _Float16* pr = myP + lr * 32;
    const v8h plo = *(const v8h*)(pr + koff);
    const v8h phi = *(const v8h*)(pr + koff + 16);
    const v16h pa = __builtin_shufflevector(plo, phi, 0,1,2,3,4,5,6,7,8,9,10,11,12,13,14,15);
    __builtin_amdgcn_wave_barrier();

    // ---- PV: 16 channel tiles from staged V ----
    #pragma unroll
    for (int t = 0; t < 16; ++t) {
      const _Float16* vr = &shV[cur][(t * 16 + lr) * 32];
      const v8h vlo = *(const v8h*)(vr + koff);
      const v8h vhi = *(const v8h*)(vr + koff + 16);
      const v16h vb = __builtin_shufflevector(vlo, vhi, 0,1,2,3,4,5,6,7,8,9,10,11,12,13,14,15);
      acc[t] = __builtin_amdgcn_wmma_f32_16x16x32_f16(false, pa, false, vb,
                                                      (short)0, acc[t], false, false);
    }

    wait_async_all();            // next buffer landed
    __syncthreads();             // everyone done with current buffer
  }

  // one-time row-sum reduction, then normalize
  float linv[8];
  #pragma unroll
  for (int i = 0; i < 8; ++i) {
    float s = rsum[i];
    s += __shfl_xor(s, 1, 32);
    s += __shfl_xor(s, 2, 32);
    s += __shfl_xor(s, 4, 32);
    s += __shfl_xor(s, 8, 32);
    linv[i] = 1.f / s;
  }

  // ---- epilogue: gamma * (acc/sum) + source; contiguous 16B stores along m ----
  const float gam = gamma_p[0];
  #pragma unroll
  for (int t = 0; t < 16; ++t) {
    const size_t base = ((size_t)b * C_ + t * 16 + lr) * N_ + m0 + 8 * half;
    v4f r0, r1;
    #pragma unroll
    for (int i = 0; i < 4; ++i) {
      r0[i] = gam * acc[t][i]     * linv[i]     + src[base + i];
      r1[i] = gam * acc[t][i + 4] * linv[i + 4] + src[base + 4 + i];
    }
    *(v4f*)(out + base)     = r0;
    *(v4f*)(out + base + 4) = r1;
  }
}

// ---------------------------------------------------------------------------
extern "C" void kernel_launch(void* const* d_in, const int* in_sizes, int n_in,
                              void* d_out, int out_size, void* d_ws, size_t ws_size,
                              hipStream_t stream) {
  const float* source = (const float*)d_in[0];
  const float* pose   = (const float*)d_in[1];
  const float* Wq     = (const float*)d_in[2];
  const float* bq     = (const float*)d_in[3];
  const float* Wk     = (const float*)d_in[4];
  const float* bk     = (const float*)d_in[5];
  const float* Wv     = (const float*)d_in[6];
  const float* bv     = (const float*)d_in[7];
  const float* gamma  = (const float*)d_in[8];
  float* out = (float*)d_out;

  // workspace: qT (1 MB) | kT (1 MB) | vF (8 MB), all f16
  _Float16* qT = (_Float16*)d_ws;
  _Float16* kT = qT + (size_t)B_ * N_ * CQ;
  _Float16* vF = kT + (size_t)B_ * N_ * CQ;

  proj_kernel<<<dim3(B_ * (N_ / 64)), dim3(256), 0, stream>>>(
      source, pose, Wq, bq, Wk, bk, Wv, bv, qT, kT, vF);

  attn_kernel<<<dim3(B_ * (N_ / 16) / WPB), dim3(256), 0, stream>>>(
      qT, kT, vF, source, gamma, out);
}
